// LSTM_AE_25391846654743
// MI455X (gfx1250) — matrix operand order repaired and checked by reading, not compiled, not run on hardware
//
#include <hip/hip_runtime.h>
#include <hip/hip_bf16.h>

typedef __attribute__((ext_vector_type(16))) _Float16 v16h;
typedef __attribute__((ext_vector_type(8)))  _Float16 v8h;
typedef __attribute__((ext_vector_type(8)))  float    v8f;

#define TSTEPS 64
#define NBATCH 1024
#define NFEAT  128

// ---------------------------------------------------------------------------
// Pack kernels
// ---------------------------------------------------------------------------

// f32 -> f16 activation pack (input x already has stride 128 == In_pad)
__global__ void pack_x_kernel(const float* __restrict__ x,
                              _Float16* __restrict__ act, int n) {
  int i = blockIdx.x * blockDim.x + threadIdx.x;
  if (i < n) act[i] = (_Float16)x[i];
}

// Pack LSTM weights into WMMA B-fragment order:
//   Bpack[ ((kb*N4 + j)*2 + sub)*16 + i ] = Wcat[j][kb*32 + (i/8)*16 + sub*8 + i%8]
__global__ void pack_w_kernel(const float* __restrict__ Wih,
                              const float* __restrict__ Whh,
                              const float* __restrict__ bih,
                              const float* __restrict__ bhh,
                              _Float16* __restrict__ bpack,
                              float* __restrict__ biasp,
                              int In, int H, int In_pad, int H_pad) {
  const int K_pad = In_pad + H_pad;
  const int N4 = 4 * H_pad;
  int idx = blockIdx.x * blockDim.x + threadIdx.x;
  if (idx >= K_pad * N4) return;
  int j = idx / K_pad;
  int k = idx % K_pad;
  int gate = j / H_pad, u = j % H_pad;
  float w = 0.f;
  if (u < H) {
    int row = gate * H + u;
    if (k < In_pad) { if (k < In) w = Wih[(size_t)row * In + k]; }
    else            { int kk = k - In_pad; if (kk < H) w = Whh[(size_t)row * H + kk]; }
  }
  int kb = k >> 5, sub = (k >> 3) & 1, e = (((k >> 4) & 1) << 3) | (k & 7);
  bpack[(((size_t)kb * N4 + j) * 2 + sub) * 16 + e] = (_Float16)w;
  if (k == 0) biasp[j] = (u < H) ? (bih[gate * H + u] + bhh[gate * H + u]) : 0.f;
}

// Pack output projection W (NFEAT x HID), K_pad=128.
__global__ void pack_outw_kernel(const float* __restrict__ W,
                                 const float* __restrict__ b,
                                 _Float16* __restrict__ bpack,
                                 float* __restrict__ biasp,
                                 int K, int K_pad, int N) {
  int idx = blockIdx.x * blockDim.x + threadIdx.x;
  if (idx >= K_pad * N) return;
  int j = idx / K_pad;
  int k = idx % K_pad;
  float w = (k < K) ? W[(size_t)j * K + k] : 0.f;
  int kb = k >> 5, sub = (k >> 3) & 1, e = (((k >> 4) & 1) << 3) | (k & 7);
  bpack[(((size_t)kb * N + j) * 2 + sub) * 16 + e] = (_Float16)w;
  if (k == 0) biasp[j] = b[j];
}

// ---------------------------------------------------------------------------
// Fused LSTM layer: persistent time loop, WMMA recurrence, async x staging.
// One workgroup owns 16 batch rows; waves tile the 4*H_pad gate width.
// Weight (B) fragments preloaded into registers once, reused for all t.
// x_{t+1} is fetched with global_load_async_to_lds (double buffered) and
// overlapped with the step-t WMMA + cell update.
// ---------------------------------------------------------------------------
template <int IN_PAD, int H_PAD>
__global__ __launch_bounds__(8 * H_PAD)
void lstm_layer_kernel(const _Float16* __restrict__ actIn,   // (T*N) x IN_PAD
                       _Float16* __restrict__ actOut,        // (T*N) x H_PAD
                       const _Float16* __restrict__ bpack,
                       const float* __restrict__ biasp) {
  constexpr int K_PAD = IN_PAD + H_PAD;
  constexpr int N4    = 4 * H_PAD;
  constexpr int KB    = K_PAD / 32;
  constexpr int XCH   = 2 * IN_PAD;      // 16B async chunks for a 16-row slab

  __shared__ __align__(16) _Float16 xstage[2][16][IN_PAD]; // double-buffered x_t
  __shared__ __align__(16) _Float16 hbuf[16][H_PAD];       // h_{t-1} (f16)
  __shared__ __align__(16) float    gbuf[16][N4];          // gate pre-activations
  __shared__ float                  cbuf[16][H_PAD];       // cell state

  const int lane = threadIdx.x;                 // 0..31 (wave32)
  const int wv   = threadIdx.y;                 // column-tile id
  const int tid  = wv * 32 + lane;
  const int nthr = 8 * H_PAD;
  const int n0   = blockIdx.x * 16;             // batch-row slab
  const int jb   = wv * 16;                     // this wave's 16 gate columns
  const int lh   = lane >> 4;                   // lane half (0/1)
  const int ll   = lane & 15;

  // Preload constant weight (B) fragments: one 32B vector per k-block.
  v16h breg[KB];
#pragma unroll
  for (int kb = 0; kb < KB; ++kb) {
    size_t off = (((size_t)kb * N4 + jb + ll) * 2 + lh) * 16;
    breg[kb] = *(const v16h*)(bpack + off);
  }
  const float bias_j = biasp[jb + ll];

  // Async stage of one 16-row x_t slab (contiguous 16*IN_PAD halves) into LDS.
  // LDS byte offset = low 32 bits of the flat shared-memory address.
  auto issue_x_async = [&](int t, int buf) {
    if (tid < XCH) {
      const _Float16* g =
          actIn + ((size_t)t * NBATCH + n0) * IN_PAD + (size_t)tid * 8;
      unsigned ldsoff =
          (unsigned)(size_t)(&xstage[buf][0][0]) + (unsigned)tid * 16u;
      asm volatile("global_load_async_to_lds_b128 %0, %1, off"
                   :: "v"(ldsoff), "v"(g)
                   : "memory");
    }
  };

  // h = c = 0
  for (int i = tid; i < 16 * H_PAD; i += nthr) {
    int m = i / H_PAD, u = i % H_PAD;
    cbuf[m][u] = 0.f;
    hbuf[m][u] = (_Float16)0.f;
  }
  // Kick off x_0 while the barrier settles.
  issue_x_async(0, 0);
  __syncthreads();

  int cur = 0;
  for (int t = 0; t < TSTEPS; ++t) {
    // Retire this wave's async x_t chunks, then make them globally visible.
    asm volatile("s_wait_asynccnt 0x0" ::: "memory");
    __syncthreads();

    // Overlap: start fetching x_{t+1} into the other buffer now.
    if (t + 1 < TSTEPS) issue_x_async(t + 1, cur ^ 1);

    // g = [x_t | h] @ Wcat   (16 x K_PAD) x (K_PAD x 16) per wave
    v8f acc = {};
#pragma unroll
    for (int kb = 0; kb < KB; ++kb) {
      union { v16h v; v8h h8[2]; } af;
      const int kk = kb * 32 + lh * 8;
      if (kb * 32 < IN_PAD) {
        af.h8[0] = *(const v8h*)&xstage[cur][ll][kk];
        af.h8[1] = *(const v8h*)&xstage[cur][ll][kk + 16];
      } else {
        const int kh = kk - IN_PAD;
        af.h8[0] = *(const v8h*)&hbuf[ll][kh];
        af.h8[1] = *(const v8h*)&hbuf[ll][kh + 16];
      }
      acc = __builtin_amdgcn_wmma_f32_16x16x32_f16(false, af.v, false, breg[kb],
                                                   (short)0, acc, false, false);
    }
    // C layout: VGPR r -> M = r + 8*lane_half, N = jb + (lane&15)
#pragma unroll
    for (int r = 0; r < 8; ++r) {
      int m = r + lh * 8;
      gbuf[m][jb + ll] = acc[r] + bias_j;
    }
    __syncthreads();

    // LSTM cell update (gates i,f,g,o are H_PAD-wide column blocks)
    for (int i = tid; i < 16 * H_PAD; i += nthr) {
      int m = i / H_PAD, u = i % H_PAD;
      float gi = gbuf[m][u];
      float gf = gbuf[m][H_PAD + u];
      float gg = gbuf[m][2 * H_PAD + u];
      float go = gbuf[m][3 * H_PAD + u];
      float si = 1.f / (1.f + __expf(-gi));
      float sf = 1.f / (1.f + __expf(-gf));
      float so = 1.f / (1.f + __expf(-go));
      float c  = sf * cbuf[m][u] + si * tanhf(gg);
      float h  = so * tanhf(c);
      cbuf[m][u] = c;
      _Float16 hh = (_Float16)h;
      hbuf[m][u] = hh;                                           // recurrent input
      actOut[((size_t)t * NBATCH + n0 + m) * H_PAD + u] = hh;    // layer output
    }
    __syncthreads();
    cur ^= 1;
  }
}

// ---------------------------------------------------------------------------
// Output projection: (65536 x 128 f16) @ (128 x 128) + bias -> f32 d_out
// ---------------------------------------------------------------------------
__global__ __launch_bounds__(256)
void out_gemm_kernel(const _Float16* __restrict__ act,
                     const _Float16* __restrict__ bpack,
                     const float* __restrict__ biasp,
                     float* __restrict__ out) {
  const int lane = threadIdx.x, wv = threadIdx.y;
  const int m0 = blockIdx.x * 16;
  const int jb = wv * 16;
  const int lh = lane >> 4, ll = lane & 15;
  v8f acc = {};
#pragma unroll
  for (int kb = 0; kb < 4; ++kb) {
    size_t boff = (((size_t)kb * NFEAT + jb + ll) * 2 + lh) * 16;
    v16h b = *(const v16h*)(bpack + boff);
    union { v16h v; v8h h8[2]; } af;
    const _Float16* arow = act + (size_t)(m0 + ll) * 128 + kb * 32 + lh * 8;
    af.h8[0] = *(const v8h*)(arow);
    af.h8[1] = *(const v8h*)(arow + 16);
    acc = __builtin_amdgcn_wmma_f32_16x16x32_f16(false, af.v, false, b,
                                                 (short)0, acc, false, false);
  }
  const float bj = biasp[jb + ll];
#pragma unroll
  for (int r = 0; r < 8; ++r) {
    int m = m0 + r + lh * 8;
    out[(size_t)m * NFEAT + jb + ll] = acc[r] + bj;
  }
}

// ---------------------------------------------------------------------------
// Host side
// ---------------------------------------------------------------------------
struct LayerDesc { int In, H; };
static const LayerDesc g_layers[10] = {
  {128,115},{115,102},{102,90},{90,77},{77,64},     // encoder
  {64,64},{64,77},{77,90},{90,102},{102,115}};      // decoder

static inline int pad32(int v) { return (v + 31) & ~31; }
static inline size_t align256(size_t v) { return (v + 255) & ~(size_t)255; }

static void launch_layer(int In_pad, int H_pad,
                         const _Float16* in, _Float16* out,
                         const _Float16* bp, const float* bb, hipStream_t s) {
  dim3 grid(NBATCH / 16);
#define LSTM_CASE(I, Hh)                                                     \
  if (In_pad == (I) && H_pad == (Hh)) {                                      \
    dim3 blk(32, (4 * (Hh)) / 16);                                           \
    lstm_layer_kernel<I, Hh><<<grid, blk, 0, s>>>(in, out, bp, bb);          \
    return;                                                                  \
  }
  LSTM_CASE(128, 128)
  LSTM_CASE(128, 96)
  LSTM_CASE(96, 96)
  LSTM_CASE(96, 64)
  LSTM_CASE(64, 64)
  LSTM_CASE(64, 96)
  LSTM_CASE(96, 128)
#undef LSTM_CASE
}

extern "C" void kernel_launch(void* const* d_in, const int* in_sizes, int n_in,
                              void* d_out, int out_size, void* d_ws, size_t ws_size,
                              hipStream_t stream) {
  (void)in_sizes; (void)n_in; (void)out_size; (void)ws_size;

  const size_t ROWS = (size_t)TSTEPS * NBATCH;   // 65536

  // Workspace layout (deterministic every call)
  char* p = (char*)d_ws;
  size_t cur = 0;
  auto alloc = [&](size_t bytes) -> char* {
    char* r = p + cur;
    cur = align256(cur + bytes);
    return r;
  };

  _Float16* actA = (_Float16*)alloc(ROWS * 128 * sizeof(_Float16));
  _Float16* actB = (_Float16*)alloc(ROWS * 128 * sizeof(_Float16));

  _Float16* bpack[10];
  float*    biasp[10];
  for (int l = 0; l < 10; ++l) {
    int In_pad = pad32(g_layers[l].In), H_pad = pad32(g_layers[l].H);
    int K_pad = In_pad + H_pad, N4 = 4 * H_pad;
    bpack[l] = (_Float16*)alloc((size_t)K_pad * N4 * sizeof(_Float16));
    biasp[l] = (float*)alloc((size_t)N4 * sizeof(float));
  }
  _Float16* bpackO = (_Float16*)alloc((size_t)128 * 128 * sizeof(_Float16));
  float*    biaspO = (float*)alloc(128 * sizeof(float));

  // ---- pack weights (all layers) ----
  for (int l = 0; l < 10; ++l) {
    int base = (l < 5) ? (1 + 4 * l) : (21 + 4 * (l - 5));
    const float* Wih = (const float*)d_in[base + 0];
    const float* Whh = (const float*)d_in[base + 1];
    const float* bih = (const float*)d_in[base + 2];
    const float* bhh = (const float*)d_in[base + 3];
    int In = g_layers[l].In, H = g_layers[l].H;
    int In_pad = pad32(In), H_pad = pad32(H);
    int total = (In_pad + H_pad) * 4 * H_pad;
    pack_w_kernel<<<(total + 255) / 256, 256, 0, stream>>>(
        Wih, Whh, bih, bhh, bpack[l], biasp[l], In, H, In_pad, H_pad);
  }
  {
    const float* outW = (const float*)d_in[41];
    const float* outb = (const float*)d_in[42];
    int total = 128 * 128;
    pack_outw_kernel<<<(total + 255) / 256, 256, 0, stream>>>(
        outW, outb, bpackO, biaspO, /*K=*/115, /*K_pad=*/128, /*N=*/128);
  }

  // ---- pack input activations ----
  {
    const float* x = (const float*)d_in[0];
    int n = (int)(ROWS * NFEAT);
    pack_x_kernel<<<(n + 255) / 256, 256, 0, stream>>>(x, actA, n);
  }

  // ---- 10 fused LSTM layers, ping-pong activation buffers ----
  _Float16* cin = actA;
  _Float16* cout = actB;
  for (int l = 0; l < 10; ++l) {
    int In_pad = pad32(g_layers[l].In), H_pad = pad32(g_layers[l].H);
    launch_layer(In_pad, H_pad, cin, cout, bpack[l], biasp[l], stream);
    _Float16* t = cin; cin = cout; cout = t;
  }

  // ---- output projection (last layer H_pad == 128) ----
  {
    dim3 grid(ROWS / 16);
    dim3 blk(32, 8);
    out_gemm_kernel<<<grid, blk, 0, stream>>>(cin, bpackO, biaspO, (float*)d_out);
  }
}
